// GNN_84482006713250
// MI455X (gfx1250) — compile-verified, matrix operand order checked by reading, not run on hardware
//
#include <hip/hip_runtime.h>

typedef __attribute__((ext_vector_type(16))) _Float16 v16h;
typedef __attribute__((ext_vector_type(8)))  _Float16 v8h;
typedef __attribute__((ext_vector_type(8)))  float    v8f;

#define WMMA_F16(a, b, c) \
    __builtin_amdgcn_wmma_f32_16x16x32_f16(false, (a), false, (b), (short)0, (c), false, false)

__device__ __forceinline__ float sigf(float x) { return 1.0f / (1.0f + __expf(-x)); }

// 16-byte-aligned LDS load of 16 halves (row stride 136 is only 16B aligned)
__device__ __forceinline__ v16h ld16(const _Float16* p) {
    v8h lo = *(const v8h*)p;
    v8h hi = *(const v8h*)(p + 8);
    return __builtin_shufflevector(lo, hi, 0,1,2,3,4,5,6,7,8,9,10,11,12,13,14,15);
}

// ---------------- pack ncf -> (2048, 512) f16, zero padded past col 384 ----------------
__global__ void build_ncf(const float* __restrict__ ld, const float* __restrict__ lk,
                          const float* __restrict__ tg, _Float16* __restrict__ ncf) {
    int idx = blockIdx.x * blockDim.x + threadIdx.x;      // 2048*512
    int row = idx >> 9, c = idx & 511;
    int b = row >> 7, i = row & 127;
    float v;
    if (c < 128)       v = ld[(b * 128 + i) * 128 + c];
    else if (c < 256)  v = lk[(b * 128 + i) * 128 + (c - 128)];
    else if (c == 256) v = tg[b * 128 + i];
    else if (c < 385)  v = ((c - 257) == i) ? 1.0f : 0.0f;
    else               v = 0.0f;
    ncf[idx] = (_Float16)v;
}

// ---------------- f32 -> f16 row-major convert with column window + zero pad ----------------
__global__ void cvt_pad(const float* __restrict__ src, int srcStride, int colOff, int srcK,
                        _Float16* __restrict__ dst, int dstK, int total) {
    int idx = blockIdx.x * blockDim.x + threadIdx.x;
    if (idx >= total) return;
    int r = idx / dstK, c = idx % dstK;
    float v = (c < srcK) ? src[(size_t)r * srcStride + colOff + c] : 0.0f;
    dst[idx] = (_Float16)v;
}

// ---------------- C(M,128) = A(M,K)f16 @ W(128,K)f16^T [+bias][+addend][SiLU] ----------------
// grid.x = M/16 row tiles; 256 threads = 8 waves; wave w owns columns [16w,16w+16)
__global__ void gemm128(const _Float16* __restrict__ A, int K,
                        const _Float16* __restrict__ W,
                        const float* __restrict__ bias,
                        const float* __restrict__ addend,
                        float* __restrict__ outF,
                        _Float16* __restrict__ outH, int act) {
    const int lane = threadIdx.x & 31, wv = threadIdx.x >> 5;
    const int m0 = blockIdx.x * 16, n0 = wv * 16;
    const int l16 = lane & 15, hh = lane >> 4;
    const _Float16* Ap = A + (size_t)(m0 + l16) * K + hh * 16;
    const _Float16* Wp = W + (size_t)(n0 + l16) * K + hh * 16;
    v8f acc = {};
    for (int k0 = 0; k0 < K; k0 += 32) {
        v16h a = *(const v16h*)(Ap + k0);
        v16h b = *(const v16h*)(Wp + k0);
        acc = WMMA_F16(a, b, acc);
    }
    const int n = n0 + l16;
    const float bn = bias ? bias[n] : 0.0f;
#pragma unroll
    for (int v = 0; v < 8; ++v) {
        int m = m0 + v + 8 * hh;
        float x = acc[v] + bn;
        if (addend) x += addend[(size_t)m * 128 + n];
        if (act == 1) x = x * sigf(x);
        outF[(size_t)m * 128 + n] = x;
        if (outH) outH[(size_t)m * 128 + n] = (_Float16)x;
    }
}

// ---------------- fused GRU over t=0..127, one block per 16-row tile ----------------
// prod[m,t,:] = SiLU(ya[m,:] + yb[b,t,:]);  gx = prod@Wih^T;  gh = h@Whh^T; gates; h update.
// Wave w owns gate columns [16w,16w+16): keeps h in D-layout accumulator registers and
// the recurrent weights Whh entirely in registers (off the serial critical path).
__global__ void gru128(const float* __restrict__ ya, const float* __restrict__ yb,
                       const _Float16* __restrict__ Wih, const _Float16* __restrict__ Whh,
                       const float* __restrict__ bih, const float* __restrict__ bhh,
                       _Float16* __restrict__ hOutH) {
    __shared__ float    ya_s[16 * 128];
    __shared__ float    yb_s[128];
    __shared__ _Float16 prod_s[16 * 136];
    __shared__ _Float16 h_s[16 * 136];

    const int tid = threadIdx.x;
    const int lane = tid & 31, wv = tid >> 5;
    const int m0 = blockIdx.x * 16, b = m0 >> 7;
    const int l16 = lane & 15, hh = lane >> 4;

    for (int idx = tid; idx < 16 * 128; idx += 256) {
        int mm = idx >> 7, cc = idx & 127;
        ya_s[idx] = ya[(size_t)(m0 + mm) * 128 + cc];
    }
    for (int idx = tid; idx < 16 * 136; idx += 256) h_s[idx] = (_Float16)0.0f;

    const int n = wv * 16 + l16;                  // hidden column owned by this lane
    const float bxr = bih[n], bxz = bih[128 + n], bxn = bih[256 + n];
    const float bhr = bhh[n], bhz = bhh[128 + n], bhn = bhh[256 + n];

    const _Float16* WihR = Wih + (size_t)(n)       * 128 + hh * 16;
    const _Float16* WihZ = Wih + (size_t)(128 + n) * 128 + hh * 16;
    const _Float16* WihN = Wih + (size_t)(256 + n) * 128 + hh * 16;
    const _Float16* WhhR = Whh + (size_t)(n)       * 128 + hh * 16;
    const _Float16* WhhZ = Whh + (size_t)(128 + n) * 128 + hh * 16;
    const _Float16* WhhN = Whh + (size_t)(256 + n) * 128 + hh * 16;

    // recurrent weights: t-invariant, keep in registers (12 x v16h = 96 VGPRs)
    v16h whr[4], whz[4], whn[4];
#pragma unroll
    for (int j = 0; j < 4; ++j) {
        whr[j] = *(const v16h*)(WhhR + 32 * j);
        whz[j] = *(const v16h*)(WhhZ + 32 * j);
        whn[j] = *(const v16h*)(WhhN + 32 * j);
    }

    const _Float16* Aprod = prod_s + l16 * 136 + hh * 16;
    const _Float16* Ahid  = h_s    + l16 * 136 + hh * 16;
    const float*    ybRow = yb + (size_t)b * 128 * 128;

    v8f h = {};

    for (int t = 0; t < 128; ++t) {
        if (tid < 128) yb_s[tid] = ybRow[t * 128 + tid];
        if (t + 1 < 128 && tid < 4)   // pull next step's 512B row toward L0 (global_prefetch_b8)
            __builtin_prefetch(ybRow + (t + 1) * 128 + tid * 32, 0, 0);
        __syncthreads();                           // yb ready; prior h_s reads done
        {   // build prod tile (16x128 f16) : 8 contiguous elements per thread
            int e = tid * 8;
            int cc = e & 127;
            _Float16* pp = prod_s + (e >> 7) * 136 + cc;
#pragma unroll
            for (int j = 0; j < 8; ++j) {
                float x = ya_s[e + j] + yb_s[cc + j];
                pp[j] = (_Float16)(x * sigf(x));
            }
        }
        __syncthreads();                           // prod ready, h_s stable

        v8f axr = {}, axz = {}, axn = {}, ahr = {}, ahz = {}, ahn = {};
#pragma unroll
        for (int j = 0; j < 4; ++j) {
            const int k0 = 32 * j;
            v16h ah = ld16(Ahid + k0);             // critical path: LDS + register weights only
            ahr = WMMA_F16(ah, whr[j], ahr);
            ahz = WMMA_F16(ah, whz[j], ahz);
            ahn = WMMA_F16(ah, whn[j], ahn);
            v16h ap = ld16(Aprod + k0);
            v16h w;
            w = *(const v16h*)(WihR + k0); axr = WMMA_F16(ap, w, axr);
            w = *(const v16h*)(WihZ + k0); axz = WMMA_F16(ap, w, axz);
            w = *(const v16h*)(WihN + k0); axn = WMMA_F16(ap, w, axn);
        }
        __syncthreads();                           // everyone done reading h_s
#pragma unroll
        for (int v = 0; v < 8; ++v) {
            float rg = sigf(axr[v] + bxr + ahr[v] + bhr);
            float zg = sigf(axz[v] + bxz + ahz[v] + bhz);
            float ng = tanhf(axn[v] + bxn + rg * (ahn[v] + bhn));
            h[v] = (1.0f - zg) * ng + zg * h[v];
            h_s[(v + 8 * hh) * 136 + n] = (_Float16)h[v];   // publish for next step
        }
    }
#pragma unroll
    for (int v = 0; v < 8; ++v)
        hOutH[(size_t)(m0 + v + 8 * hh) * 128 + n] = (_Float16)h[v];
}

// ---------------- distances = r' . dr_W + dr_b : one wave32 per row, butterfly reduce ----------
__global__ void dr_head(const float* __restrict__ rp, const float* __restrict__ drW,
                        const float* __restrict__ drb, float* __restrict__ out) {
    int gid  = blockIdx.x * blockDim.x + threadIdx.x;
    int row  = gid >> 5;                       // one wave per row
    int lane = threadIdx.x & 31;
    if (row >= 2048) return;
    const float* r = rp + (size_t)row * 128;
    float s = 0.0f;
#pragma unroll
    for (int j = 0; j < 4; ++j) s += r[lane + 32 * j] * drW[lane + 32 * j];
#pragma unroll
    for (int off = 16; off > 0; off >>= 1) s += __shfl_xor(s, off, 32);
    if (lane == 0) out[row] = s + drb[0];
}

extern "C" void kernel_launch(void* const* d_in, const int* in_sizes, int n_in,
                              void* d_out, int out_size, void* d_ws, size_t ws_size,
                              hipStream_t stream) {
    (void)in_sizes; (void)n_in; (void)out_size; (void)ws_size;
    const float* ld    = (const float*)d_in[0];
    const float* lk    = (const float*)d_in[1];
    const float* tg    = (const float*)d_in[2];
    const float* biW   = (const float*)d_in[3];
    const float* bib   = (const float*)d_in[4];
    const float* buW   = (const float*)d_in[5];
    const float* bub   = (const float*)d_in[6];
    const float* Wih   = (const float*)d_in[7];
    const float* Whh   = (const float*)d_in[8];
    const float* bih   = (const float*)d_in[9];
    const float* bhh   = (const float*)d_in[10];
    const float* pmW   = (const float*)d_in[11];
    const float* pmb   = (const float*)d_in[12];
    const float* dfW   = (const float*)d_in[13];
    const float* dfb   = (const float*)d_in[14];
    const float* drW   = (const float*)d_in[15];
    const float* drb   = (const float*)d_in[16];

    char* ws = (char*)d_ws;
    size_t off = 0;
    auto take = [&](size_t bytes) { char* p = ws + off; off = (off + bytes + 255) & ~(size_t)255; return p; };

    _Float16* ncf_h = (_Float16*)take(2048 * 512 * 2);
    _Float16* biWh  = (_Float16*)take(128 * 512 * 2);
    _Float16* WaNh  = (_Float16*)take(128 * 512 * 2);
    _Float16* WbNh  = (_Float16*)take(128 * 512 * 2);
    _Float16* WaRh  = (_Float16*)take(128 * 128 * 2);
    _Float16* WbRh  = (_Float16*)take(128 * 128 * 2);
    _Float16* pmWh  = (_Float16*)take(128 * 128 * 2);
    _Float16* dfWh  = (_Float16*)take(128 * 128 * 2);
    _Float16* Wihh  = (_Float16*)take(384 * 128 * 2);
    _Float16* Whhh  = (_Float16*)take(384 * 128 * 2);
    float*    rpF   = (float*)   take(2048 * 128 * 4);
    _Float16* rpH   = (_Float16*)take(2048 * 128 * 2);
    float*    yaC   = (float*)   take(2048 * 128 * 4);
    float*    ybC   = (float*)   take(2048 * 128 * 4);
    float*    yaB   = (float*)   take(2048 * 128 * 4);
    float*    ybB   = (float*)   take(2048 * 128 * 4);
    _Float16* hH    = (_Float16*)take(2048 * 128 * 2);

    float* outDist = (float*)d_out;              // (2048,128)
    float* outDr   = outDist + 2048 * 128;       // (2048,1)

    // pack / convert (once per call; deterministic)
    build_ncf<<<(2048 * 512) / 256, 256, 0, stream>>>(ld, lk, tg, ncf_h);
    cvt_pad<<<256, 256, 0, stream>>>(biW, 385, 0, 385, biWh, 512, 128 * 512);
    cvt_pad<<<256, 256, 0, stream>>>(buW, 1026, 128, 385, WaNh, 512, 128 * 512);  // Wa ncf part
    cvt_pad<<<256, 256, 0, stream>>>(buW, 1026, 641, 385, WbNh, 512, 128 * 512);  // Wb ncf part
    cvt_pad<<<64, 256, 0, stream>>>(buW, 1026, 0,   128, WaRh, 128, 128 * 128);   // Wa r' part
    cvt_pad<<<64, 256, 0, stream>>>(buW, 1026, 513, 128, WbRh, 128, 128 * 128);   // Wb r' part
    cvt_pad<<<64, 256, 0, stream>>>(pmW, 128, 0, 128, pmWh, 128, 128 * 128);
    cvt_pad<<<64, 256, 0, stream>>>(dfW, 128, 0, 128, dfWh, 128, 128 * 128);
    cvt_pad<<<192, 256, 0, stream>>>(Wih, 128, 0, 128, Wihh, 128, 384 * 128);
    cvt_pad<<<192, 256, 0, stream>>>(Whh, 128, 0, 128, Whhh, 128, 384 * 128);

    // r' = SiLU(ncf @ biW^T + b);  step-invariant ya/yb parts from ncf
    gemm128<<<128, 256, 0, stream>>>(ncf_h, 512, biWh, bib, nullptr, rpF, rpH, 1);
    gemm128<<<128, 256, 0, stream>>>(ncf_h, 512, WaNh, nullptr, nullptr, yaC, nullptr, 0);
    gemm128<<<128, 256, 0, stream>>>(ncf_h, 512, WbNh, bub, nullptr, ybC, nullptr, 0);

    for (int s = 0; s < 2; ++s) {
        gemm128<<<128, 256, 0, stream>>>(rpH, 128, WaRh, nullptr, yaC, yaB, nullptr, 0);
        gemm128<<<128, 256, 0, stream>>>(rpH, 128, WbRh, nullptr, ybC, ybB, nullptr, 0);
        gru128<<<128, 256, 0, stream>>>(yaB, ybB, Wihh, Whhh, bih, bhh, hH);
        gemm128<<<128, 256, 0, stream>>>(hH, 128, pmWh, pmb, nullptr, rpF, rpH, 0);
    }

    gemm128<<<128, 256, 0, stream>>>(rpH, 128, dfWh, dfb, nullptr, outDist, nullptr, 0);
    dr_head<<<8, 256, 0, stream>>>(rpF, drW, drb, outDr);
}
